// SelfAttentionModule_40510131536516
// MI455X (gfx1250) — matrix operand order repaired and checked by reading, not compiled, not run on hardware
//
#include <hip/hip_runtime.h>

// ---------------------------------------------------------------------------
// Self-attention without softmax:  Out = Q (K^T V), per batch.
//   X:[4,4096,1024] f32, W*:[128,1024], b*:[128]
// Strategy: bf16 WMMA (v_wmma_f32_16x16x32_bf16) with f32 accumulation.
// B fragments are processed in two fenced groups of 4: load clauses with
// incremental waits and back-to-back wmmas, WITHOUT spilling (peak ~130 VGPR).
// ---------------------------------------------------------------------------

typedef __attribute__((ext_vector_type(16))) __bf16       v16bf;
typedef __attribute__((ext_vector_type(8)))  float        v8f;
typedef __attribute__((ext_vector_type(4)))  float        v4f;
typedef __attribute__((ext_vector_type(4)))  unsigned int v4u;

#define BATCH 4
#define SEQ   4096
#define EMB   1024
#define DK    128
#define CHUNKS 16
#define JCHUNK (SEQ / CHUNKS)   // 256
#define QPAD  (DK + 8)          // LDS stride: 68 dwords -> conflict-free transpose read

// POD fragment holder: 8 VGPRs = 16 bf16 per lane (A/B operand of 16x16x32)
union BF16Frag {
    v4u    u[2];   // two 16-byte halves
    v16bf  v;
    __bf16 e[16];
};

__device__ __forceinline__ v8f wmma_bf16(const BF16Frag& a, const BF16Frag& b, v8f c) {
    return __builtin_amdgcn_wmma_f32_16x16x32_bf16(false, a.v, false, b.v,
                                                   (short)0, c, false, false);
}

// ---------------------------------------------------------------------------
// Kernel 0: convert the three weight matrices to bf16 (row-major; L2-resident).
// ---------------------------------------------------------------------------
__global__ void cvt_weights(const float* __restrict__ Wq,
                            const float* __restrict__ Wk,
                            const float* __restrict__ Wv,
                            __bf16* __restrict__ Wbf) {
    const int per = DK * EMB;
    int idx = blockIdx.x * blockDim.x + threadIdx.x;       // 0 .. 3*per
    const float* src = (idx < per) ? Wq : (idx < 2 * per) ? Wk : Wv;
    int off = (idx < per) ? idx : (idx < 2 * per) ? idx - per : idx - 2 * per;
    Wbf[idx] = (__bf16)src[off];
}

// ---------------------------------------------------------------------------
// Kernel 1: projections.  Grid (SEQ/16, BATCH), block 96 (= 3 waves).
// Wave w in {0,1,2} computes a 16x128 tile of Q / K / V for its row tile.
//   - Q: staged through LDS (transpose) then stored row-major bf16
//   - K, V: stored transposed bf16 ([128][4096]); the D-fragment is
//     j-contiguous per lane so this is one b128 store per N-tile.
// ---------------------------------------------------------------------------
__global__ void proj_kernel(const float* __restrict__ X,
                            const float* __restrict__ bq,
                            const float* __restrict__ bk,
                            const float* __restrict__ bv,
                            const __bf16* __restrict__ Wbf,
                            __bf16* __restrict__ Qbf,
                            __bf16* __restrict__ Kt,
                            __bf16* __restrict__ Vt) {
    __shared__ __bf16 qstage[16 * QPAD];     // used only by wave 0 (Q)

    const int batch   = blockIdx.y;
    const int rowbase = blockIdx.x * 16;
    const int w       = threadIdx.x >> 5;      // 0=Q 1=K 2=V (wave-uniform)
    const int lane    = threadIdx.x & 31;
    const int lrow    = lane & 15;
    const int hi      = lane >> 4;

    const float*  bias = (w == 0) ? bq : (w == 1) ? bk : bv;
    const __bf16* Wm   = Wbf + (size_t)w * DK * EMB;
    const float*  Xrow = X + ((size_t)batch * SEQ + rowbase + lrow) * EMB;

    v8f acc[8] = {};

    for (int ks = 0; ks < EMB / 32; ++ks) {          // 32 k-steps of 32
        const int c0 = ks * 32 + hi * 16;

        // ---- issue A loads (X, f32) ----
        v4f f0 = *(const v4f*)(Xrow + c0);
        v4f f1 = *(const v4f*)(Xrow + c0 + 4);
        v4f f2 = *(const v4f*)(Xrow + c0 + 8);
        v4f f3 = *(const v4f*)(Xrow + c0 + 12);

        // prefetch next k-step's X data (global_prefetch_b8)
        __builtin_prefetch(Xrow + c0 + 32, 0, 1);

        // ---- group 0: B fragments 0..3 ----
        BF16Frag b0[4];
        #pragma unroll
        for (int nt = 0; nt < 4; ++nt) {
            const __bf16* bp = Wm + (size_t)(nt * 16 + lrow) * EMB + ks * 32 + hi * 16;
            b0[nt].u[0] = *(const v4u*)(bp);
            b0[nt].u[1] = *(const v4u*)(bp + 8);
        }

        // ---- convert A while group-0 B loads are in flight ----
        BF16Frag a;
        a.e[0]  = (__bf16)f0.x; a.e[1]  = (__bf16)f0.y; a.e[2]  = (__bf16)f0.z; a.e[3]  = (__bf16)f0.w;
        a.e[4]  = (__bf16)f1.x; a.e[5]  = (__bf16)f1.y; a.e[6]  = (__bf16)f1.z; a.e[7]  = (__bf16)f1.w;
        a.e[8]  = (__bf16)f2.x; a.e[9]  = (__bf16)f2.y; a.e[10] = (__bf16)f2.z; a.e[11] = (__bf16)f2.w;
        a.e[12] = (__bf16)f3.x; a.e[13] = (__bf16)f3.y; a.e[14] = (__bf16)f3.z; a.e[15] = (__bf16)f3.w;

        __builtin_amdgcn_sched_barrier(0);

        #pragma unroll
        for (int nt = 0; nt < 4; ++nt)
            acc[nt] = wmma_bf16(a, b0[nt], acc[nt]);

        __builtin_amdgcn_sched_barrier(0);

        // ---- group 1: B fragments 4..7 (issue while group-0 wmmas drain) ----
        BF16Frag b1[4];
        #pragma unroll
        for (int nt = 0; nt < 4; ++nt) {
            const __bf16* bp = Wm + (size_t)((nt + 4) * 16 + lrow) * EMB + ks * 32 + hi * 16;
            b1[nt].u[0] = *(const v4u*)(bp);
            b1[nt].u[1] = *(const v4u*)(bp + 8);
        }

        __builtin_amdgcn_sched_barrier(0);

        #pragma unroll
        for (int nt = 0; nt < 4; ++nt)
            acc[nt + 4] = wmma_bf16(a, b1[nt], acc[nt + 4]);
    }

    if (w == 0) {
        // Q: write D-fragments into LDS, then read back row-major and store
        // coalesced b128s to global.
        #pragma unroll
        for (int nt = 0; nt < 8; ++nt) {
            float bb = bias[nt * 16 + lrow];
            #pragma unroll
            for (int r = 0; r < 8; ++r)
                qstage[(r + 8 * hi) * QPAD + nt * 16 + lrow] = (__bf16)(acc[nt][r] + bb);
        }
        // lane (lrow, hi) handles row lrow, 64-col half 'hi': 8 x b128 each.
        const __bf16* src = qstage + lrow * QPAD + hi * 64;
        __bf16* dst = Qbf + ((size_t)batch * SEQ + rowbase + lrow) * DK + hi * 64;
        #pragma unroll
        for (int c = 0; c < 8; ++c)
            *(v4u*)(dst + c * 8) = *(const v4u*)(src + c * 8);
    } else {
        // K / V: transposed bf16 [batch][128][seq]
        __bf16* T = (w == 1) ? Kt : Vt;
        #pragma unroll
        for (int nt = 0; nt < 8; ++nt) {
            float bb = bias[nt * 16 + lrow];
            BF16Frag t;
            #pragma unroll
            for (int r = 0; r < 8; ++r) t.e[r] = (__bf16)(acc[nt][r] + bb);
            // lane's 8 values are j = rowbase + 8*hi + (0..7): contiguous.
            __bf16* dst = T + ((size_t)batch * DK + nt * 16 + lrow) * SEQ
                            + rowbase + 8 * hi;
            *(v4u*)dst = t.u[0];
        }
    }
}

// ---------------------------------------------------------------------------
// Kernel 2: partial S = K^T V per (batch, j-chunk of 256).
// Grid (CHUNKS, BATCH), block 256 (8 waves).  Wave w owns S row-tile at=w.
// ---------------------------------------------------------------------------
__global__ void ktv_kernel(const __bf16* __restrict__ Kt,
                           const __bf16* __restrict__ Vt,
                           float* __restrict__ Sp) {
    const int batch = blockIdx.y;
    const int chunk = blockIdx.x;
    const int at    = threadIdx.x >> 5;
    const int lane  = threadIdx.x & 31;
    const int lrow  = lane & 15;
    const int hi    = lane >> 4;
    const int jbase = chunk * JCHUNK;

    const __bf16* Arow = Kt + ((size_t)batch * DK + at * 16 + lrow) * SEQ;

    v8f acc[8] = {};
    for (int ks = 0; ks < JCHUNK / 32; ++ks) {
        const int j0 = jbase + ks * 32;

        BF16Frag a;
        a.u[0] = *(const v4u*)(Arow + j0 + 16 * hi);
        a.u[1] = *(const v4u*)(Arow + j0 + 8 + 16 * hi);

        BF16Frag b0[4];
        #pragma unroll
        for (int bt = 0; bt < 4; ++bt) {
            const __bf16* bp = Vt + ((size_t)batch * DK + bt * 16 + lrow) * SEQ
                                  + j0 + 16 * hi;
            b0[bt].u[0] = *(const v4u*)(bp);
            b0[bt].u[1] = *(const v4u*)(bp + 8);
        }

        __builtin_amdgcn_sched_barrier(0);

        #pragma unroll
        for (int bt = 0; bt < 4; ++bt)
            acc[bt] = wmma_bf16(a, b0[bt], acc[bt]);

        __builtin_amdgcn_sched_barrier(0);

        BF16Frag b1[4];
        #pragma unroll
        for (int bt = 0; bt < 4; ++bt) {
            const __bf16* bp = Vt + ((size_t)batch * DK + (bt + 4) * 16 + lrow) * SEQ
                                  + j0 + 16 * hi;
            b1[bt].u[0] = *(const v4u*)(bp);
            b1[bt].u[1] = *(const v4u*)(bp + 8);
        }

        __builtin_amdgcn_sched_barrier(0);

        #pragma unroll
        for (int bt = 0; bt < 4; ++bt)
            acc[bt + 4] = wmma_bf16(a, b1[bt], acc[bt + 4]);
    }

    float* out = Sp + ((size_t)batch * CHUNKS + chunk) * DK * DK;
    #pragma unroll
    for (int bt = 0; bt < 8; ++bt) {
        #pragma unroll
        for (int r = 0; r < 8; ++r) {
            int ai = at * 16 + r + 8 * hi;
            int bi = bt * 16 + lrow;
            out[ai * DK + bi] = acc[bt][r];
        }
    }
}

// ---------------------------------------------------------------------------
// Kernel 2b: reduce the partial S matrices and transpose to St[v][a] bf16.
// ---------------------------------------------------------------------------
__global__ void reduce_s(const float* __restrict__ Sp, __bf16* __restrict__ St) {
    int idx   = blockIdx.x * blockDim.x + threadIdx.x;   // 4*128*128
    int a     = idx & 127;
    int v     = (idx >> 7) & 127;
    int batch = idx >> 14;
    float s = 0.f;
    for (int c = 0; c < CHUNKS; ++c)
        s += Sp[(((size_t)batch * CHUNKS + c) * DK + a) * DK + v];
    St[((size_t)batch * DK + v) * DK + a] = (__bf16)s;
}

// ---------------------------------------------------------------------------
// Kernel 3: Out = Q @ S.  Grid (SEQ/64, BATCH), block 128 (4 waves).
// ---------------------------------------------------------------------------
__global__ void out_kernel(const __bf16* __restrict__ Qbf,
                           const __bf16* __restrict__ St,
                           float* __restrict__ Out) {
    const int batch   = blockIdx.y;
    const int w       = threadIdx.x >> 5;
    const int rowbase = blockIdx.x * 64 + w * 16;
    const int lane    = threadIdx.x & 31;
    const int lrow    = lane & 15;
    const int hi      = lane >> 4;

    const __bf16* Arow = Qbf + ((size_t)batch * SEQ + rowbase + lrow) * DK;

    v8f acc[8] = {};
    #pragma unroll
    for (int ks = 0; ks < DK / 32; ++ks) {
        const int k0 = ks * 32;

        BF16Frag a;
        a.u[0] = *(const v4u*)(Arow + k0 + 16 * hi);
        a.u[1] = *(const v4u*)(Arow + k0 + 8 + 16 * hi);

        BF16Frag b0[4];
        #pragma unroll
        for (int bt = 0; bt < 4; ++bt) {
            const __bf16* bp = St + ((size_t)batch * DK + bt * 16 + lrow) * DK
                                  + k0 + 16 * hi;
            b0[bt].u[0] = *(const v4u*)(bp);
            b0[bt].u[1] = *(const v4u*)(bp + 8);
        }

        __builtin_amdgcn_sched_barrier(0);

        #pragma unroll
        for (int bt = 0; bt < 4; ++bt)
            acc[bt] = wmma_bf16(a, b0[bt], acc[bt]);

        __builtin_amdgcn_sched_barrier(0);

        BF16Frag b1[4];
        #pragma unroll
        for (int bt = 0; bt < 4; ++bt) {
            const __bf16* bp = St + ((size_t)batch * DK + (bt + 4) * 16 + lrow) * DK
                                  + k0 + 16 * hi;
            b1[bt].u[0] = *(const v4u*)(bp);
            b1[bt].u[1] = *(const v4u*)(bp + 8);
        }

        __builtin_amdgcn_sched_barrier(0);

        #pragma unroll
        for (int bt = 0; bt < 4; ++bt)
            acc[bt + 4] = wmma_bf16(a, b1[bt], acc[bt + 4]);
    }

    #pragma unroll
    for (int bt = 0; bt < 8; ++bt) {
        #pragma unroll
        for (int r = 0; r < 8; ++r) {
            int row = rowbase + r + 8 * hi;
            Out[((size_t)batch * SEQ + row) * DK + bt * 16 + lrow] = acc[bt][r];
        }
    }
}

// ---------------------------------------------------------------------------
extern "C" void kernel_launch(void* const* d_in, const int* in_sizes, int n_in,
                              void* d_out, int out_size, void* d_ws, size_t ws_size,
                              hipStream_t stream) {
    (void)in_sizes; (void)n_in; (void)out_size; (void)ws_size;

    const float* X  = (const float*)d_in[0];
    const float* Wq = (const float*)d_in[1];
    const float* bq = (const float*)d_in[2];
    const float* Wk = (const float*)d_in[3];
    const float* bk = (const float*)d_in[4];
    const float* Wv = (const float*)d_in[5];
    const float* bv = (const float*)d_in[6];
    float* Out = (float*)d_out;

    char* ws = (char*)d_ws;
    const size_t MiB = 1ull << 20;
    __bf16* Qbf = (__bf16*)(ws + 0 * MiB);    // 4 MiB
    __bf16* Kt  = (__bf16*)(ws + 4 * MiB);    // 4 MiB
    __bf16* Vt  = (__bf16*)(ws + 8 * MiB);    // 4 MiB
    __bf16* Wbf = (__bf16*)(ws + 12 * MiB);   // 0.75 MiB
    float*  Sp  = (float*) (ws + 13 * MiB);   // 4 MiB
    __bf16* St  = (__bf16*)(ws + 17 * MiB);   // 128 KiB

    cvt_weights<<<dim3(3 * DK * EMB / 256), dim3(256), 0, stream>>>(Wq, Wk, Wv, Wbf);
    proj_kernel<<<dim3(SEQ / 16, BATCH), dim3(96), 0, stream>>>(X, bq, bk, bv,
                                                                Wbf, Qbf, Kt, Vt);
    ktv_kernel<<<dim3(CHUNKS, BATCH), dim3(256), 0, stream>>>(Kt, Vt, Sp);
    reduce_s<<<dim3(BATCH * DK * DK / 256), dim3(256), 0, stream>>>(Sp, St);
    out_kernel<<<dim3(SEQ / 64, BATCH), dim3(128), 0, stream>>>(Qbf, St, Out);
}
